// BiDirectionalAttention_19207093748074
// MI455X (gfx1250) — matrix operand order repaired and checked by reading, not compiled, not run on hardware
//
#include <hip/hip_runtime.h>
#include <hip/hip_bf16.h>

typedef __attribute__((ext_vector_type(16))) _Float16 v16h;
typedef __attribute__((ext_vector_type(8)))  _Float16 v8h;
typedef __attribute__((ext_vector_type(8)))  float    v8f;

#define NEG_INF_F (-1e30f)

constexpr int B  = 64;
constexpr int LV = 1024;
constexpr int LQ = 64;
constexpr int D  = 512;

// ---------------------------------------------------------------------------
// K0: vw[b,v] = QDF[b,v,:]·w_v   and   qw[b,q] = query[b,q,:]·w_q
// One wave32 per row; coalesced strided reads + shuffle reduction.
// ---------------------------------------------------------------------------
__global__ __launch_bounds__(256)
void k0_rowdots(const float* __restrict__ qdf,
                const float* __restrict__ query,
                const float* __restrict__ simw,
                float* __restrict__ vw,
                float* __restrict__ qw) {
  int wave = (int)((blockIdx.x * blockDim.x + threadIdx.x) >> 5);
  int lane = threadIdx.x & 31;
  const int nVW = B * LV;
  const int nQW = B * LQ;
  const float* row;
  const float* w;
  float* outp;
  int oidx;
  if (wave < nVW) {
    row = qdf + (size_t)wave * D; w = simw;     outp = vw; oidx = wave;
  } else if (wave < nVW + nQW) {
    int r = wave - nVW;
    row = query + (size_t)r * D;  w = simw + D; outp = qw; oidx = r;
  } else {
    return;
  }
  float acc = 0.f;
  #pragma unroll 4
  for (int i = 0; i < D; i += 32) {
    int d = i + lane;
    acc = fmaf(row[d], w[d], acc);
  }
  #pragma unroll
  for (int off = 16; off > 0; off >>= 1) acc += __shfl_xor(acc, off, 32);
  if (lane == 0) outp[oidx] = acc;
}

// ---------------------------------------------------------------------------
// K1: per (batch, 64-row v-tile):
//   GEMM1: sim(64x64) = (QDF*w_p) @ query^T   [WMMA f16, K=512, dbl-buffered]
//   bias + mask + fp32 softmax over q, row-max -> m
//   GEMM2: V2Q(64x512) = a @ query            [WMMA f16, K=64]
//   LDS-bounced float4 epilogue writes QAL sections 1 (V2Q), 2 (QDF*V2Q)
// ---------------------------------------------------------------------------
__global__ __launch_bounds__(256)
void k1_main(const float* __restrict__ qdf,
             const float* __restrict__ query,
             const float* __restrict__ vmask,
             const float* __restrict__ qmask,
             const float* __restrict__ simw,
             const float* __restrict__ vw,
             const float* __restrict__ qw,
             float* __restrict__ m_out,
             float* __restrict__ out) {
  __shared__ alignas(16) _Float16 q16[LQ * D];       // 64 KB query tile f16
  __shared__ alignas(16) float    sim[64 * 64];      // 16 KB logits scratch
  __shared__ alignas(16) _Float16 aT[64 * 64];       //  8 KB softmax weights
  __shared__ alignas(16) _Float16 stg[2][64 * 32];   //  8 KB dbl-buffered A
  __shared__ alignas(16) float    vout[8][16 * 64];  // 32 KB per-wave bounce

  const int b     = blockIdx.x / (LV / 64);
  const int vtile = blockIdx.x % (LV / 64);
  const int vbase = vtile * 64;
  const int tid   = threadIdx.x;
  const int wv    = tid >> 5;           // wave id 0..7
  const int lane  = tid & 31;
  const int lrow  = lane & 15;          // M / N selector within tile
  const int hi    = lane >> 4;          // lane half
  const int kbA   = hi ? 16 : 0;        // A-frag K base (contiguous k = 16*hi + e)
  const int kbB   = hi ? 16 : 0;        // B-frag K base (contiguous k = 16*hi + e, same as A)

  const float* wp = simw + 2 * D;
  const float* Q  = query + (size_t)b * LQ * D;
  const float* V  = qdf   + (size_t)b * LV * D;

  // stage query tile -> LDS f16: float4 loads, packed b128 stores
  #pragma unroll
  for (int it = 0; it < 16; ++it) {
    int idx = (it * 256 + tid) * 8;
    const float4* qp = (const float4*)(Q + idx);
    float4 x0 = qp[0], x1 = qp[1];
    v8h h;
    h[0] = (_Float16)x0.x; h[1] = (_Float16)x0.y;
    h[2] = (_Float16)x0.z; h[3] = (_Float16)x0.w;
    h[4] = (_Float16)x1.x; h[5] = (_Float16)x1.y;
    h[6] = (_Float16)x1.z; h[7] = (_Float16)x1.w;
    *(v8h*)&q16[idx] = h;
  }

  // stage one 64x32 chunk of QDF*w_p -> stg[buf] (one b128 per thread)
  auto stageA = [&](int buf, int k0) {
    int idx = tid * 8;                  // 2048 halves = 256 threads * 8
    int r = idx >> 5, c = idx & 31;
    const float4* vp = (const float4*)(V + (size_t)(vbase + r) * D + k0 + c);
    const float4* wq = (const float4*)(wp + k0 + c);
    float4 x0 = vp[0], x1 = vp[1];
    float4 w0 = wq[0], w1 = wq[1];
    v8h h;
    h[0] = (_Float16)(x0.x * w0.x); h[1] = (_Float16)(x0.y * w0.y);
    h[2] = (_Float16)(x0.z * w0.z); h[3] = (_Float16)(x0.w * w0.w);
    h[4] = (_Float16)(x1.x * w1.x); h[5] = (_Float16)(x1.y * w1.y);
    h[6] = (_Float16)(x1.z * w1.z); h[7] = (_Float16)(x1.w * w1.w);
    *(v8h*)&stg[buf][idx] = h;
  };

  // ---------------- GEMM1: sim = (QDF*w_p) @ query^T ----------------
  {
    const int mtile = wv & 3;           // 4 row tiles
    const int nhalf = wv >> 2;          // 2 col halves (2 tiles each)
    v8f acc0 = {};
    v8f acc1 = {};
    const int q0 = (nhalf * 2 + 0) * 16 + lrow;
    const int q1 = (nhalf * 2 + 1) * 16 + lrow;
    const int arow = mtile * 16 + lrow;

    stageA(0, 0);
    int buf = 0;
    for (int k0 = 0; k0 < D; k0 += 32) {
      __syncthreads();                  // stg[buf] ready, prev reads done
      if (k0 + 32 < D) stageA(buf ^ 1, k0 + 32);  // overlap next chunk

      v16h af, bf0, bf1;
      #pragma unroll
      for (int e = 0; e < 8; ++e) {
        af[e]     = stg[buf][arow * 32 + kbA + e];
        af[e + 8] = stg[buf][arow * 32 + kbA + 8 + e];
      }
      #pragma unroll
      for (int e = 0; e < 16; ++e) {
        bf0[e] = q16[q0 * D + k0 + kbB + e];
        bf1[e] = q16[q1 * D + k0 + kbB + e];
      }
      acc0 = __builtin_amdgcn_wmma_f32_16x16x32_f16(false, af, false, bf0,
                                                    (short)0, acc0, false, false);
      acc1 = __builtin_amdgcn_wmma_f32_16x16x32_f16(false, af, false, bf1,
                                                    (short)0, acc1, false, false);
      buf ^= 1;
    }
    // C/D layout: VGPR j -> M = j + 8*hi, N = lane&15
    const int rbase = mtile * 16 + hi * 8;
    #pragma unroll
    for (int j = 0; j < 8; ++j) {
      sim[(rbase + j) * 64 + (nhalf * 2 + 0) * 16 + lrow] = acc0[j];
      sim[(rbase + j) * 64 + (nhalf * 2 + 1) * 16 + lrow] = acc1[j];
    }
  }
  __syncthreads();

  // ---------------- bias + mask + softmax over q (fp32) ----------------
  if (tid < 64) {
    const int r = tid;
    const int v = vbase + r;
    const float rowadd = vw[b * LV + v];
    const float vm     = vmask[b * LV + v];
    float mx = -3.4e38f;
    for (int q = 0; q < 64; ++q) {
      float l = sim[r * 64 + q] + rowadd + qw[b * LQ + q]
              + (1.0f - vm * qmask[b * LQ + q]) * NEG_INF_F;
      sim[r * 64 + q] = l;
      mx = fmaxf(mx, l);
    }
    m_out[b * LV + v] = mx;             // row-max feeds Q2V softmax
    float s = 0.f;
    for (int q = 0; q < 64; ++q) {
      float e = __expf(sim[r * 64 + q] - mx);
      sim[r * 64 + q] = e;
      s += e;
    }
    const float inv = 1.0f / s;
    for (int q = 0; q < 64; ++q) aT[r * 64 + q] = (_Float16)(sim[r * 64 + q] * inv);
  }
  __syncthreads();

  // ---------------- GEMM2: V2Q = a @ query ----------------
  // wave wv owns d in [wv*64, wv*64+64). Preload all 8 strided B-frags once.
  {
    v16h bfr0[4], bfr1[4];
    #pragma unroll
    for (int nt = 0; nt < 4; ++nt) {
      const int d = wv * 64 + nt * 16 + lrow;
      #pragma unroll
      for (int e = 0; e < 16; ++e) {
        bfr0[nt][e] = q16[(kbB + e) * D + d];
        bfr1[nt][e] = q16[(32 + kbB + e) * D + d];
      }
    }
    float* vo = vout[wv];

    for (int mtile = 0; mtile < 4; ++mtile) {
      v16h a0, a1;
      const int arow = mtile * 16 + lrow;
      #pragma unroll
      for (int e = 0; e < 8; ++e) {
        a0[e]     = aT[arow * 64 + (kbA + e)];
        a0[e + 8] = aT[arow * 64 + (kbA + 8 + e)];
        a1[e]     = aT[arow * 64 + (32 + kbA + e)];
        a1[e + 8] = aT[arow * 64 + (32 + kbA + 8 + e)];
      }
      #pragma unroll
      for (int nt = 0; nt < 4; ++nt) {
        v8f acc = {};
        acc = __builtin_amdgcn_wmma_f32_16x16x32_f16(false, a0, false, bfr0[nt],
                                                     (short)0, acc, false, false);
        acc = __builtin_amdgcn_wmma_f32_16x16x32_f16(false, a1, false, bfr1[nt],
                                                     (short)0, acc, false, false);
        #pragma unroll
        for (int j = 0; j < 8; ++j)
          vo[(hi * 8 + j) * 64 + nt * 16 + lrow] = acc[j];
      }
      // per-wave LDS bounce -> 256B-contiguous float4 global stores
      // (same-wave DS ops are in-order; no barrier needed)
      #pragma unroll
      for (int rr = 0; rr < 8; ++rr) {
        const int r = rr * 2 + hi;
        const int c = lrow * 4;
        const float4 av = *(const float4*)&vo[r * 64 + c];
        const int v = vbase + mtile * 16 + r;
        const int d = wv * 64 + c;
        const float4 qd = *(const float4*)(V + (size_t)v * D + d);
        const size_t o = ((size_t)(b * LV + v)) * (size_t)(4 * D);
        *(float4*)(out + o + D + d) = av;
        float4 pr;
        pr.x = qd.x * av.x; pr.y = qd.y * av.y;
        pr.z = qd.z * av.z; pr.w = qd.w * av.w;
        *(float4*)(out + o + 2 * D + d) = pr;
      }
    }
  }
}

// ---------------------------------------------------------------------------
// K2: per batch: bw = softmax_v(m);  Q2V[b,d] = sum_v bw[v]*QDF[b,v,d]
// ---------------------------------------------------------------------------
__global__ __launch_bounds__(256)
void k2_q2v(const float* __restrict__ qdf,
            const float* __restrict__ m_in,
            float* __restrict__ q2v) {
  __shared__ float bw[LV];
  __shared__ float red[256];
  const int b   = blockIdx.x;
  const int tid = threadIdx.x;
  const float* m = m_in + (size_t)b * LV;

  float mx = -3.4e38f;
  for (int v = tid; v < LV; v += 256) mx = fmaxf(mx, m[v]);
  red[tid] = mx; __syncthreads();
  for (int s = 128; s > 0; s >>= 1) {
    if (tid < s) red[tid] = fmaxf(red[tid], red[tid + s]);
    __syncthreads();
  }
  mx = red[0]; __syncthreads();

  float sum = 0.f;
  for (int v = tid; v < LV; v += 256) {
    float e = __expf(m[v] - mx);
    bw[v] = e;
    sum += e;
  }
  red[tid] = sum; __syncthreads();
  for (int s = 128; s > 0; s >>= 1) {
    if (tid < s) red[tid] += red[tid + s];
    __syncthreads();
  }
  const float inv = 1.0f / red[0];
  for (int v = tid; v < LV; v += 256) bw[v] *= inv;
  __syncthreads();

  // float2 per thread: one pass covers all D, full-row coalescing per v
  const float2* V2 = (const float2*)(qdf + (size_t)b * LV * D);
  float2 acc; acc.x = 0.f; acc.y = 0.f;
  #pragma unroll 4
  for (int v = 0; v < LV; ++v) {
    const float w = bw[v];
    const float2 x = V2[(size_t)v * (D / 2) + tid];
    acc.x = fmaf(w, x.x, acc.x);
    acc.y = fmaf(w, x.y, acc.y);
  }
  *(float2*)(q2v + (size_t)b * D + 2 * tid) = acc;
}

// ---------------------------------------------------------------------------
// K3: float4 coalesced writer for sections 0 (QDF) and 3 (QDF*Q2V)
// ---------------------------------------------------------------------------
__global__ __launch_bounds__(256)
void k3_tail(const float* __restrict__ qdf,
             const float* __restrict__ q2v,
             float* __restrict__ out) {
  size_t i4 = ((size_t)blockIdx.x * 256 + threadIdx.x) * 4;
  if (i4 >= (size_t)B * LV * D) return;
  const int d     = (int)(i4 % D);
  const size_t bv = i4 / D;
  const int b     = (int)(bv / LV);
  const float4 qd = *(const float4*)(qdf + i4);
  const float4 qv = *(const float4*)(q2v + (size_t)b * D + d);
  const size_t o  = bv * (size_t)(4 * D);
  *(float4*)(out + o + d) = qd;
  float4 pr;
  pr.x = qd.x * qv.x; pr.y = qd.y * qv.y;
  pr.z = qd.z * qv.z; pr.w = qd.w * qv.w;
  *(float4*)(out + o + 3 * D + d) = pr;
}

// ---------------------------------------------------------------------------
extern "C" void kernel_launch(void* const* d_in, const int* in_sizes, int n_in,
                              void* d_out, int out_size, void* d_ws, size_t ws_size,
                              hipStream_t stream) {
  const float* qdf   = (const float*)d_in[0];
  const float* query = (const float*)d_in[1];
  const float* vmask = (const float*)d_in[2];
  const float* qmask = (const float*)d_in[3];
  const float* simw  = (const float*)d_in[4];
  float* out = (float*)d_out;

  float* ws  = (float*)d_ws;
  float* vw  = ws;                       // B*LV
  float* qw  = vw + B * LV;              // B*LQ
  float* m   = qw + B * LQ;              // B*LV
  float* q2v = m + B * LV;               // B*D

  // K0: (B*LV + B*LQ) waves, 8 waves/block
  const int rows = B * LV + B * LQ;      // 69632, multiple of 8
  k0_rowdots<<<dim3(rows / 8), dim3(256), 0, stream>>>(qdf, query, simw, vw, qw);

  // K1: one block per (batch, 64-row v-tile)
  k1_main<<<dim3(B * (LV / 64)), dim3(256), 0, stream>>>(
      qdf, query, vmask, qmask, simw, vw, qw, m, out);

  // K2: one block per batch
  k2_q2v<<<dim3(B), dim3(256), 0, stream>>>(qdf, m, q2v);

  // K3: float4 tail over B*LV*D
  k3_tail<<<dim3((B * LV * D) / 4 / 256), dim3(256), 0, stream>>>(qdf, q2v, out);
}
